// ConvTXCDR_41686952575011
// MI455X (gfx1250) — matrix operand order repaired and checked by reading, not compiled, not run on hardware
//
#include <hip/hip_runtime.h>
#include <hip/hip_bf16.h>

typedef __attribute__((ext_vector_type(16))) _Float16 v16h;
typedef __attribute__((ext_vector_type(8)))  float    v8f;

#define B_    32
#define T_    16
#define DIN   768
#define DSAE  16384
#define K_    32
#define KTOT  2304            // 3*DIN, GEMM K dim (c = 3*d + tap, natural conv_w order)
#define KSTEPS 72             // 2304 / 32
#define TD    12288           // T*DIN

// ---------------- K0: zero the sparse-z output region ----------------
__global__ void k_zero(float* __restrict__ z, int n) {
    int i = blockIdx.x * blockDim.x + threadIdx.x;
    if (i < n) z[i] = 0.0f;
}

// ---------------- K1: build f16 A-matrix, pre-swizzled into WMMA A-fragment layout ----
// A[b, c] with c = 3d+tap:  tap0 -> Xsum - x[T-1],  tap1 -> Xsum,  tap2 -> Xsum - x[0]
// Storage: Af[((kk*2 + mtile)*32 + lane)*16 + elem], per ISA 16-bit A 16x32 layout:
//   lane l (<16):  elems 0..7 = K 0..7,  elems 8..15 = K 16..23
//   lane l+16:     elems 0..7 = K 8..15, elems 8..15 = K 24..31
__global__ void k_build_a(const float* __restrict__ x, _Float16* __restrict__ Af) {
    int gid = blockIdx.x * 256 + threadIdx.x;          // 0 .. 32*768-1
    if (gid >= B_ * DIN) return;
    int b = gid / DIN, d = gid % DIN;
    const float* xp = x + (size_t)b * TD + d;
    float s = 0.0f;
#pragma unroll
    for (int t = 0; t < T_; ++t) s += xp[t * DIN];
    float a3[3] = { s - xp[(T_ - 1) * DIN], s, s - xp[0] };
    int m = b >> 4, l = b & 15;
#pragma unroll
    for (int tap = 0; tap < 3; ++tap) {
        int c  = 3 * d + tap;
        int kk = c >> 5, kl = c & 31;
        int lane, e;
        if      (kl <  8) { lane = l;      e = kl;      }
        else if (kl < 16) { lane = l + 16; e = kl - 8;  }
        else if (kl < 24) { lane = l;      e = kl - 8;  }
        else              { lane = l + 16; e = kl - 16; }
        Af[(size_t)((kk * 2 + m) * 32 + lane) * 16 + e] = (_Float16)a3[tap];
    }
}

// ---------------- K2: encode GEMM via v_wmma_f32_16x16x32_f16 ----------------
// One wave per 16-column N-tile; computes both 16-row M-tiles (conv_w read once).
// Lane (n, half): column s = stile*16+n; B elems 0..7 = conv_w[s, c0 .. c0+7], elems 8..15 = conv_w[s, c0+16 .. c0+23],
// c0 = kk*32 + half*8  -> lane pair (n, n+16) covers one 128B cacheline per step.
__global__ void __launch_bounds__(128) k_encode(const float* __restrict__ conv_w,
                                                const float* __restrict__ b_enc,
                                                const _Float16* __restrict__ Af,
                                                float* __restrict__ pre) {
    int wave  = threadIdx.x >> 5;
    int lane  = threadIdx.x & 31;
    int stile = blockIdx.x * 4 + wave;                 // 0..1023
    int n     = lane & 15;
    int half  = lane >> 4;
    int s     = stile * 16 + n;

    const float*    wp = conv_w + (size_t)s * KTOT + half * 8;
    const _Float16* ap = Af + lane * 16;

    v8f acc0 = {};
    v8f acc1 = {};

    for (int kk = 0; kk < KSTEPS; ++kk) {
        const float4* w4 = (const float4*)wp;
        float4 w0 = w4[0], w1 = w4[1], w2 = w4[4], w3 = w4[5];
        v16h bf;
        bf[0]  = (_Float16)w0.x; bf[1]  = (_Float16)w0.y; bf[2]  = (_Float16)w0.z; bf[3]  = (_Float16)w0.w;
        bf[4]  = (_Float16)w1.x; bf[5]  = (_Float16)w1.y; bf[6]  = (_Float16)w1.z; bf[7]  = (_Float16)w1.w;
        bf[8]  = (_Float16)w2.x; bf[9]  = (_Float16)w2.y; bf[10] = (_Float16)w2.z; bf[11] = (_Float16)w2.w;
        bf[12] = (_Float16)w3.x; bf[13] = (_Float16)w3.y; bf[14] = (_Float16)w3.z; bf[15] = (_Float16)w3.w;

        v16h a0 = *(const v16h*)(ap);          // M-tile 0 fragment
        v16h a1 = *(const v16h*)(ap + 512);    // M-tile 1 fragment

        acc0 = __builtin_amdgcn_wmma_f32_16x16x32_f16(false, a0, false, bf, (short)0, acc0, false, false);
        acc1 = __builtin_amdgcn_wmma_f32_16x16x32_f16(false, a1, false, bf, (short)0, acc1, false, false);

        wp += 32;
        ap += 1024;
        __builtin_prefetch(wp + 128, 0, 0);    // stream conv_w ahead (global_prefetch)
    }

    // C layout: lane n holds (M=r, N=n) r=0..7; lane n+16 holds M=r+8. pre is [b][s].
    float be = b_enc[s];
    float* p = pre + (size_t)(half * 8) * DSAE + s;
#pragma unroll
    for (int r = 0; r < 8; ++r) {
        p[(size_t)r * DSAE]        = acc0[r] + be;   // b = half*8 + r
        p[(size_t)(r + 16) * DSAE] = acc1[r] + be;   // b = 16 + half*8 + r
    }
}

// ---------------- K3: top-K=32 per row (ties -> lowest index, like lax.top_k) ----------------
__global__ void __launch_bounds__(256) k_topk(const float* __restrict__ pre,
                                              float* __restrict__ topv, int* __restrict__ topi,
                                              float* __restrict__ z) {
    extern __shared__ unsigned char smem_raw[];
    float* vals = (float*)smem_raw;                       // 16384 floats (dynamic LDS: 64KB+)
    float* rv   = (float*)(smem_raw + DSAE * 4);          // 256 floats
    int*   ri   = (int*)  (smem_raw + DSAE * 4 + 1024);   // 256 ints

    int b = blockIdx.x, tid = threadIdx.x;
    const float* row = pre + (size_t)b * DSAE;
    for (int i = tid; i < DSAE; i += 256) vals[i] = row[i];
    __syncthreads();

    for (int it = 0; it < K_; ++it) {
        float bv = -__builtin_inff();
        int   bi = 0x7fffffff;
        for (int i = tid; i < DSAE; i += 256) {           // ascending -> keeps lowest idx on tie
            float v = vals[i];
            if (v > bv) { bv = v; bi = i; }
        }
        rv[tid] = bv; ri[tid] = bi;
        __syncthreads();
        for (int off = 128; off > 0; off >>= 1) {
            if (tid < off) {
                float ov = rv[tid + off]; int oi = ri[tid + off];
                if (ov > rv[tid] || (ov == rv[tid] && oi < ri[tid])) { rv[tid] = ov; ri[tid] = oi; }
            }
            __syncthreads();
        }
        if (tid == 0) {
            int bidx = ri[0]; float bval = rv[0];
            topv[b * K_ + it] = bval;
            topi[b * K_ + it] = bidx;
            z[(size_t)b * DSAE + bidx] = bval > 0.0f ? bval : 0.0f;
            vals[bidx] = -__builtin_inff();
        }
        __syncthreads();
    }
}

// ---------------- K4: sparse decode + per-block squared-error partials ----------------
__global__ void __launch_bounds__(256) k_decode(const float* __restrict__ W_dec,
                                                const float* __restrict__ b_dec,
                                                const float* __restrict__ x,
                                                const float* __restrict__ topv,
                                                const int* __restrict__ topi,
                                                float* __restrict__ xhat,
                                                float* __restrict__ partial) {
    __shared__ float zv[K_];
    __shared__ int   zi[K_];
    __shared__ float red[256];

    int b  = blockIdx.x / 48;                 // 12288/256 = 48 blocks per batch row
    int td = (blockIdx.x % 48) * 256 + threadIdx.x;
    if (threadIdx.x < K_) {
        float v = topv[b * K_ + threadIdx.x];
        zv[threadIdx.x] = v > 0.0f ? v : 0.0f;        // relu -> z value
        zi[threadIdx.x] = topi[b * K_ + threadIdx.x];
    }
    __syncthreads();

    float acc = b_dec[td];
#pragma unroll 8
    for (int j = 0; j < K_; ++j)
        acc += zv[j] * W_dec[(size_t)zi[j] * TD + td];

    xhat[(size_t)b * TD + td] = acc;
    float diff = acc - x[(size_t)b * TD + td];
    red[threadIdx.x] = diff * diff;
    __syncthreads();
    for (int off = 128; off > 0; off >>= 1) {
        if (threadIdx.x < off) red[threadIdx.x] += red[threadIdx.x + off];
        __syncthreads();
    }
    if (threadIdx.x == 0) partial[blockIdx.x] = red[0];
}

// ---------------- K5: deterministic loss reduction ----------------
__global__ void __launch_bounds__(256) k_loss(const float* __restrict__ partial, float* __restrict__ out) {
    __shared__ float red[256];
    float s = 0.0f;
    for (int i = threadIdx.x; i < 1536; i += 256) s += partial[i];
    red[threadIdx.x] = s;
    __syncthreads();
    for (int off = 128; off > 0; off >>= 1) {
        if (threadIdx.x < off) red[threadIdx.x] += red[threadIdx.x + off];
        __syncthreads();
    }
    if (threadIdx.x == 0) out[0] = red[0] / (float)(B_ * T_);   // mean over (b,t)
}

extern "C" void kernel_launch(void* const* d_in, const int* in_sizes, int n_in,
                              void* d_out, int out_size, void* d_ws, size_t ws_size,
                              hipStream_t stream) {
    const float* x      = (const float*)d_in[0];   // (32,16,768)
    const float* conv_w = (const float*)d_in[1];   // (16384,768,3)
    const float* b_enc  = (const float*)d_in[2];   // (16384,)
    const float* W_dec  = (const float*)d_in[3];   // (16384,16,768)
    const float* b_dec  = (const float*)d_in[4];   // (16,768)

    float* out      = (float*)d_out;
    float* out_loss = out;                         // 1
    float* out_xhat = out + 1;                     // 32*16*768 = 393216
    float* out_z    = out + 1 + (size_t)B_ * TD;   // 32*16384  = 524288

    unsigned char* ws = (unsigned char*)d_ws;
    float*     pre     = (float*)ws;                                  // 2 MB
    _Float16*  Af      = (_Float16*)(ws + (size_t)B_ * DSAE * 4);     // 144 KB
    float*     topv    = (float*)(ws + 2097152 + 196608);
    int*       topi    = (int*)  (ws + 2097152 + 196608 + 4096);
    float*     partial = (float*)(ws + 2097152 + 196608 + 8192);      // 1536 floats

    k_zero   <<<2048, 256, 0, stream>>>(out_z, B_ * DSAE);
    k_build_a<<<96,   256, 0, stream>>>(x, Af);
    k_encode <<<256,  128, 0, stream>>>(conv_w, b_enc, Af, pre);
    k_topk   <<<32,   256, DSAE * 4 + 2048, stream>>>(pre, topv, topi, out_z);
    k_decode <<<1536, 256, 0, stream>>>(W_dec, b_dec, x, topv, topi, out_xhat, partial);
    k_loss   <<<1,    256, 0, stream>>>(partial, out_loss);
}
